// SwinBlock1D_14577119002778
// MI455X (gfx1250) — matrix operand order repaired and checked by reading, not compiled, not run on hardware
//
#include <hip/hip_runtime.h>
#include <hip/hip_bf16.h>
#include <math.h>

typedef __bf16 bf16;
typedef __attribute__((ext_vector_type(16))) __bf16 bf16x16;
typedef __attribute__((ext_vector_type(8)))  float  f32x8;

#define BB 8
#define CC 256
#define LL 16384
#define NH 8
#define HD 32
#define WSZ 8
#define SHIFT_ 4
#define MT 64            // tokens per block tile
#define ASTR 264         // lnA / ao LDS row stride (bf16 elems)
#define QSTR 776         // qkv LDS row stride (bf16 elems)
#define YSTR 261         // y LDS row stride (f32, odd -> conflict-free transpose)
#define HSTR 72          // hbuf LDS row stride (bf16 elems)
#define XSTR 66          // kernel A tile stride (f32)

// ---- CDNA5 async global->LDS staging (ASYNCcnt-tracked, no VGPR data path) ----
__device__ __forceinline__ void async_ld_lds_b128(void* lds, const void* g) {
  unsigned lo = (unsigned)(uintptr_t)lds;  // wave-relative LDS byte address
  asm volatile("global_load_async_to_lds_b128 %0, %1, off"
               :: "v"(lo), "v"((unsigned long long)(uintptr_t)g)
               : "memory");
}
__device__ __forceinline__ void wait_async0() {
  asm volatile("s_wait_asynccnt 0x0" ::: "memory");
}

// ---- WMMA fragment helpers (CDNA5 16x16x32 bf16, wave32) ----
__device__ __forceinline__ bf16x16 frag_a(const bf16* base, int ld, int lane, int k0) {
  const bf16* row = base + (lane & 15) * ld + k0 + ((lane >> 4) << 3);
  union { bf16x16 v; unsigned u[8]; } r;
#pragma unroll
  for (int i = 0; i < 4; ++i) r.u[i]     = *(const unsigned*)(row + 2 * i);
#pragma unroll
  for (int i = 0; i < 4; ++i) r.u[4 + i] = *(const unsigned*)(row + 16 + 2 * i);
  return r.v;
}
__device__ __forceinline__ bf16x16 frag_b(const bf16* wt, int ld, int n0, int k0, int lane) {
  const bf16* row = wt + (size_t)(n0 + (lane & 15)) * ld + k0 + ((lane >> 4) << 3);
  union { bf16x16 v; unsigned u[8]; } r;
#pragma unroll
  for (int i = 0; i < 8; ++i) r.u[i] = *(const unsigned*)(row + 2 * i + ((i >> 2) << 3));
  return r.v;
}
__device__ __forceinline__ f32x8 wmma_bf16(bf16x16 a, bf16x16 b, f32x8 c) {
  return __builtin_amdgcn_wmma_f32_16x16x32_bf16(false, a, false, b, (short)0, c, false, false);
}

// ---- weight transpose + bf16 convert: dst[n*K + k] = src[k*N + n] ----
__global__ void __launch_bounds__(256) k_wconv(const float* __restrict__ src,
                                               bf16* __restrict__ dst, int K, int N) {
  int idx = blockIdx.x * 256 + threadIdx.x;
  if (idx < K * N) {
    int k = idx / N, n = idx % N;
    dst[(size_t)n * K + k] = (bf16)src[idx];
  }
}

// ---- transpose-in + LayerNorm1 ----
__global__ void __launch_bounds__(256) k_ln1(const float* __restrict__ x,
                                             const float* __restrict__ n1w,
                                             const float* __restrict__ n1b,
                                             float* __restrict__ xt,
                                             bf16* __restrict__ ln1) {
  extern __shared__ __align__(16) char smem[];
  float* tile = (float*)smem;            // CC x XSTR
  float* mu   = tile + CC * XSTR;        // MT
  float* rs   = mu + MT;
  int tid = threadIdx.x;
  int b = blockIdx.x / (LL / MT);
  int g = blockIdx.x % (LL / MT);
  int l0 = g * MT;
  const float* xb = x + (size_t)b * CC * LL;
  for (int idx = tid; idx < CC * MT; idx += 256) {
    int c = idx >> 6, i = idx & 63;
    tile[c * XSTR + i] = xb[(size_t)c * LL + l0 + i];
  }
  __syncthreads();
  if (tid < MT) {
    float s = 0.f, s2 = 0.f;
    for (int c = 0; c < CC; ++c) { float v = tile[c * XSTR + tid]; s += v; s2 += v * v; }
    float m = s * (1.f / CC);
    mu[tid] = m;
    rs[tid] = rsqrtf(s2 * (1.f / CC) - m * m + 1e-5f);
  }
  __syncthreads();
  float* xtb = xt  + ((size_t)b * LL + l0) * CC;
  bf16*  lnb = ln1 + ((size_t)b * LL + l0) * CC;
  for (int it = 0; it < MT; ++it) {
    int c = tid;
    float v = tile[c * XSTR + it];
    xtb[(size_t)it * CC + c] = v;
    lnb[(size_t)it * CC + c] = (bf16)((v - mu[it]) * rs[it] * n1w[c] + n1b[c]);
  }
}

// ---- fused: QKV GEMM -> window attention -> proj GEMM + residual -> LayerNorm2 ----
__global__ void __launch_bounds__(256) k_attn(const bf16* __restrict__ ln1,
                                              float* __restrict__ xt,
                                              const bf16* __restrict__ qkvT,
                                              const bf16* __restrict__ projT,
                                              const float* __restrict__ proj_b,
                                              const float* __restrict__ n2w,
                                              const float* __restrict__ n2b,
                                              bf16* __restrict__ ln2) {
  extern __shared__ __align__(16) char smem[];
  bf16*  lnA  = (bf16*)smem;                                   // MT x ASTR (reused as attn-out)
  bf16*  qkv  = (bf16*)(smem + (size_t)MT * ASTR * 2);         // MT x QSTR
  float* ylds = (float*)(smem + (size_t)MT * ASTR * 2 + (size_t)MT * QSTR * 2); // MT x YSTR
  float* mu   = ylds + MT * YSTR;
  float* rs   = mu + MT;
  int tid = threadIdx.x, lane = tid & 31, wv = tid >> 5;
  int b = blockIdx.x / (LL / MT);
  int g = blockIdx.x % (LL / MT);
  int base = g * MT + SHIFT_;                       // shifted-window origin (wraps mod L)
  const bf16* lng = ln1 + (size_t)b * LL * CC;
  // phase 1: async-stage 64 shifted LN1 rows into LDS (16B chunks)
  for (int idx = tid; idx < MT * (CC / 8); idx += 256) {
    int i = idx >> 5, c8 = idx & 31;
    int l = (base + i) & (LL - 1);
    async_ld_lds_b128(lnA + i * ASTR + c8 * 8, lng + (size_t)l * CC + c8 * 8);
  }
  wait_async0();
  __syncthreads();
  // phase 2: QKV = lnA(64x256) @ qkvT^T ; A fragments hoisted, reused across 6 n-tiles
  for (int mt = 0; mt < 4; ++mt) {
    bf16x16 aF[8];
#pragma unroll
    for (int ks = 0; ks < 8; ++ks) aF[ks] = frag_a(lnA + mt * 16 * ASTR, ASTR, lane, ks * 32);
    for (int jn = 0; jn < 6; ++jn) {
      int nt = wv * 6 + jn;
      f32x8 acc = {0.f,0.f,0.f,0.f,0.f,0.f,0.f,0.f};
#pragma unroll
      for (int ks = 0; ks < 8; ++ks)
        acc = wmma_bf16(aF[ks], frag_b(qkvT, CC, nt * 16, ks * 32, lane), acc);
      int n  = nt * 16 + (lane & 15);
      int mb = mt * 16 + ((lane >> 4) << 3);
#pragma unroll
      for (int r = 0; r < 8; ++r) qkv[(mb + r) * QSTR + n] = (bf16)acc[r];
    }
  }
  __syncthreads();
  // phase 3: 8 windows x 8 heads of 8x8 attention (one (win,head) per thread)
  if (tid < 64) {
    int win = tid >> 3, h = tid & 7;
    const bf16* qp = qkv + (win * WSZ) * QSTR + h * HD;
    const bf16* kp = qp + CC;
    const bf16* vp = qp + 2 * CC;
    const float scale = 0.17677669529663687f;   // 1/sqrt(32)
    for (int i = 0; i < WSZ; ++i) {
      float q[HD];
#pragma unroll
      for (int d = 0; d < HD; ++d) q[d] = (float)qp[i * QSTR + d];
      float s[WSZ], mx = -1e30f;
#pragma unroll
      for (int j = 0; j < WSZ; ++j) {
        float dot = 0.f;
#pragma unroll
        for (int d = 0; d < HD; ++d) dot += q[d] * (float)kp[j * QSTR + d];
        s[j] = dot * scale;
        mx = fmaxf(mx, s[j]);
      }
      float sum = 0.f;
#pragma unroll
      for (int j = 0; j < WSZ; ++j) { s[j] = __expf(s[j] - mx); sum += s[j]; }
      float inv = 1.f / sum;
      for (int d = 0; d < HD; ++d) {
        float o = 0.f;
#pragma unroll
        for (int j = 0; j < WSZ; ++j) o += s[j] * (float)vp[j * QSTR + d];
        lnA[(win * WSZ + i) * ASTR + h * HD + d] = (bf16)(o * inv);  // attn-out over lnA
      }
    }
  }
  __syncthreads();
  // phase 4: proj GEMM + bias + residual -> y (LDS); one mt per wave pair, A hoisted
  float* xtb = xt + (size_t)b * LL * CC;
  {
    int mt = wv >> 1;
    bf16x16 aF[8];
#pragma unroll
    for (int ks = 0; ks < 8; ++ks) aF[ks] = frag_a(lnA + mt * 16 * ASTR, ASTR, lane, ks * 32);
    for (int jn = 0; jn < 8; ++jn) {
      int nt = ((wv & 1) << 3) + jn;
      f32x8 acc = {0.f,0.f,0.f,0.f,0.f,0.f,0.f,0.f};
#pragma unroll
      for (int ks = 0; ks < 8; ++ks)
        acc = wmma_bf16(aF[ks], frag_b(projT, CC, nt * 16, ks * 32, lane), acc);
      int n  = nt * 16 + (lane & 15);
      int mb = mt * 16 + ((lane >> 4) << 3);
      float pb = proj_b[n];
#pragma unroll
      for (int r = 0; r < 8; ++r) {
        int m = mb + r;
        int l = (base + m) & (LL - 1);
        ylds[m * YSTR + n] = acc[r] + pb + xtb[(size_t)l * CC + n];
      }
    }
  }
  __syncthreads();
  // phase 5: LayerNorm2 + coalesced writes of y (in-place over xt) and ln2 (bf16)
  if (tid < MT) {
    float s = 0.f, s2 = 0.f;
    for (int c = 0; c < CC; ++c) { float v = ylds[tid * YSTR + c]; s += v; s2 += v * v; }
    float m = s * (1.f / CC);
    mu[tid] = m;
    rs[tid] = rsqrtf(s2 * (1.f / CC) - m * m + 1e-5f);
  }
  __syncthreads();
  bf16* ln2b = ln2 + (size_t)b * LL * CC;
  for (int it = 0; it < MT; ++it) {
    int c = tid;
    int l = (base + it) & (LL - 1);
    float v = ylds[it * YSTR + c];
    xtb[(size_t)l * CC + c]  = v;
    ln2b[(size_t)l * CC + c] = (bf16)((v - mu[it]) * rs[it] * n2w[c] + n2b[c]);
  }
}

// ---- fused MLP: out = y + gelu(ln2@ff1 + b1)@ff2 + b2, with transpose-out ----
__global__ void __launch_bounds__(256) k_ffn(const bf16* __restrict__ ln2,
                                             const float* __restrict__ y,
                                             const bf16* __restrict__ ff1T,
                                             const float* __restrict__ ff1b,
                                             const bf16* __restrict__ ff2T,
                                             const float* __restrict__ ff2b,
                                             float* __restrict__ out) {
  extern __shared__ __align__(16) char smem[];
  bf16*  lnA  = (bf16*)smem;                                   // MT x ASTR
  bf16*  hbuf = (bf16*)(smem + (size_t)MT * ASTR * 2);         // MT x HSTR
  float* ybuf = (float*)(smem + (size_t)MT * ASTR * 2 + (size_t)MT * HSTR * 2); // MT x YSTR
  int tid = threadIdx.x, lane = tid & 31, wv = tid >> 5;
  int b = blockIdx.x / (LL / MT);
  int g = blockIdx.x % (LL / MT);
  int l0 = g * MT;
  const bf16* lng = ln2 + ((size_t)b * LL + l0) * CC;
  for (int idx = tid; idx < MT * (CC / 8); idx += 256) {
    int i = idx >> 5, c8 = idx & 31;
    async_ld_lds_b128(lnA + i * ASTR + c8 * 8, lng + (size_t)i * CC + c8 * 8);
  }
  wait_async0();
  __syncthreads();
  f32x8 acc[8];
#pragma unroll
  for (int j = 0; j < 8; ++j) acc[j] = (f32x8){0.f,0.f,0.f,0.f,0.f,0.f,0.f,0.f};
  int ntb = wv * 2;                              // this wave's out n-tile base
  int mth = wv >> 1;                             // h-chunk m-tile for this wave
  for (int kc = 0; kc < 16; ++kc) {              // hidden dim in 64-wide chunks
    if (lane == 0 && kc < 15)
      __builtin_prefetch(ff1T + (size_t)(kc + 1) * 64 * CC, 0, 1);
    // h chunk: 4x4 tiles, 2 per wave (same mt); A fragments hoisted
    {
      bf16x16 aF[8];
#pragma unroll
      for (int ks = 0; ks < 8; ++ks) aF[ks] = frag_a(lnA + mth * 16 * ASTR, ASTR, lane, ks * 32);
#pragma unroll
      for (int j = 0; j < 2; ++j) {
        int ntc = ((wv & 1) << 1) | j;
        f32x8 h = {0.f,0.f,0.f,0.f,0.f,0.f,0.f,0.f};
#pragma unroll
        for (int ks = 0; ks < 8; ++ks)
          h = wmma_bf16(aF[ks], frag_b(ff1T, CC, kc * 64 + ntc * 16, ks * 32, lane), h);
        int ng   = kc * 64 + ntc * 16 + (lane & 15);
        float bi = ff1b[ng];
        int n    = ntc * 16 + (lane & 15);
        int mb   = mth * 16 + ((lane >> 4) << 3);
#pragma unroll
        for (int r = 0; r < 8; ++r) {
          float xv = h[r] + bi;
          hbuf[(mb + r) * HSTR + n] = (bf16)(0.5f * xv * (1.f + erff(xv * 0.70710678118f)));
        }
      }
    }
    __syncthreads();
    // out += h(64x64) @ ff2_chunk(64x256); persistent register accumulators
#pragma unroll
    for (int mt2 = 0; mt2 < 4; ++mt2) {
#pragma unroll
      for (int ks = 0; ks < 2; ++ks) {
        bf16x16 a = frag_a(hbuf + mt2 * 16 * HSTR, HSTR, lane, ks * 32);
#pragma unroll
        for (int j = 0; j < 2; ++j) {
          bf16x16 bb = frag_b(ff2T, 4 * CC, (ntb + j) * 16, kc * 64 + ks * 32, lane);
          acc[mt2 * 2 + j] = wmma_bf16(a, bb, acc[mt2 * 2 + j]);
        }
      }
    }
    __syncthreads();
  }
  // epilogue: + bias + residual, stage to LDS
  const float* yb = y + ((size_t)b * LL + l0) * CC;
#pragma unroll
  for (int mt2 = 0; mt2 < 4; ++mt2) {
#pragma unroll
    for (int j = 0; j < 2; ++j) {
      int n  = (ntb + j) * 16 + (lane & 15);
      int mb = mt2 * 16 + ((lane >> 4) << 3);
      float bi = ff2b[n];
      f32x8 a = acc[mt2 * 2 + j];
#pragma unroll
      for (int r = 0; r < 8; ++r) {
        int m = mb + r;
        ybuf[m * YSTR + n] = a[r] + bi + yb[(size_t)m * CC + n];
      }
    }
  }
  __syncthreads();
  // transpose-out: out[b][c][l0+i] = ybuf[i][c] (odd YSTR -> conflict-free columns)
  float* ob = out + (size_t)b * CC * LL + l0;
  for (int idx = tid; idx < CC * MT; idx += 256) {
    int c = idx >> 6, i = idx & 63;
    ob[(size_t)c * LL + i] = ybuf[i * YSTR + c];
  }
}

extern "C" void kernel_launch(void* const* d_in, const int* in_sizes, int n_in,
                              void* d_out, int out_size, void* d_ws, size_t ws_size,
                              hipStream_t stream) {
  const float* x      = (const float*)d_in[0];
  const float* n1w    = (const float*)d_in[1];
  const float* n1b    = (const float*)d_in[2];
  const float* n2w    = (const float*)d_in[3];
  const float* n2b    = (const float*)d_in[4];
  const float* qkv_w  = (const float*)d_in[5];
  const float* proj_w = (const float*)d_in[6];
  const float* proj_b = (const float*)d_in[7];
  const float* ff1_w  = (const float*)d_in[8];
  const float* ff1_b  = (const float*)d_in[9];
  const float* ff2_w  = (const float*)d_in[10];
  const float* ff2_b  = (const float*)d_in[11];

  char* ws = (char*)d_ws;
  size_t off = 0;
  float* xt   = (float*)(ws + off); off += (size_t)BB * LL * CC * 4;   // residual / y (in-place)
  bf16*  ln1  = (bf16*)(ws + off);  off += (size_t)BB * LL * CC * 2;
  bf16*  ln2  = (bf16*)(ws + off);  off += (size_t)BB * LL * CC * 2;
  bf16*  qkvT = (bf16*)(ws + off);  off += (size_t)3 * CC * CC * 2;
  bf16*  projT= (bf16*)(ws + off);  off += (size_t)CC * CC * 2;
  bf16*  ff1T = (bf16*)(ws + off);  off += (size_t)4 * CC * CC * 2;
  bf16*  ff2T = (bf16*)(ws + off);  off += (size_t)4 * CC * CC * 2;

  dim3 blk(256);
  k_wconv<<<(CC * 3 * CC + 255) / 256, blk, 0, stream>>>(qkv_w, qkvT, CC, 3 * CC);
  k_wconv<<<(CC * CC + 255) / 256,     blk, 0, stream>>>(proj_w, projT, CC, CC);
  k_wconv<<<(CC * 4 * CC + 255) / 256, blk, 0, stream>>>(ff1_w, ff1T, CC, 4 * CC);
  k_wconv<<<(4 * CC * CC + 255) / 256, blk, 0, stream>>>(ff2_w, ff2T, 4 * CC, CC);

  int nblk = BB * (LL / MT);
  size_t shA = (size_t)CC * XSTR * 4 + 2 * MT * 4;
  k_ln1<<<nblk, blk, shA, stream>>>(x, n1w, n1b, xt, ln1);
  size_t shB = (size_t)MT * ASTR * 2 + (size_t)MT * QSTR * 2 + (size_t)MT * YSTR * 4 + 2 * MT * 4;
  k_attn<<<nblk, blk, shB, stream>>>(ln1, xt, qkvT, projT, proj_b, n2w, n2b, ln2);
  size_t shC = (size_t)MT * ASTR * 2 + (size_t)MT * HSTR * 2 + (size_t)MT * YSTR * 4;
  k_ffn<<<nblk, blk, shC, stream>>>(ln2, xt, ff1T, ff1_b, ff2T, ff2_b, (float*)d_out);
}
